// DendroMatrixNN_19301583028815
// MI455X (gfx1250) — matrix operand order repaired and checked by reading, not compiled, not run on hardware
//
#include <hip/hip_runtime.h>

// ---------------------------------------------------------------------------
// Problem constants (from the reference)
// ---------------------------------------------------------------------------
#define BB_   4096   // batch
#define FF_   256    // features
#define L0_   128    // hidden
#define L1_   16     // out
#define EE_   2048   // edges (GEMM K)
#define NN_   2048   // nodes (GEMM N)

typedef __attribute__((ext_vector_type(16))) __bf16 v16bf;
typedef __attribute__((ext_vector_type(8)))  float  v8f;
typedef __attribute__((ext_vector_type(2)))  float  v2f;
typedef __attribute__((ext_vector_type(2)))  __bf16 v2bf;

union AFrag { uint4 q[2]; v16bf v; };
union BFrag { uint4 q[2]; v16bf v; };
union CFrag { v8f v; float f[8]; float4 q[2]; };

// pack two f32 -> one dword of two bf16 (lo = x, hi = y); lets clang pick the
// hardware v_cvt_pk_bf16_f32 when available, software RNE otherwise.
__device__ __forceinline__ unsigned pack_bf16x2(float x, float y) {
    v2f  in = {x, y};
    v2bf t  = __builtin_convertvector(in, v2bf);
    return __builtin_bit_cast(unsigned, t);
}

// ---------------------------------------------------------------------------
// Kernel 1: pack path_mat (E x N, f32) into B-FRAGMENT order:
//   P4[(((c*2 + hw)*N + n)*8 + v] = { bf16(P[K, n]), bf16(P[K+1, n]) }
//   with K = c*32 + hw*16 + 2v   (c = K-chunk, hw = half-wave, v = VGPR idx)
// A wave's 16-wide B fragment is then two contiguous b128 loads.
// ---------------------------------------------------------------------------
__global__ __launch_bounds__(256) void pack_path_kernel(
    const float* __restrict__ pm, unsigned* __restrict__ P4)
{
    const int n  = blockIdx.x * 256 + threadIdx.x;   // 0..N-1
    const int kp = blockIdx.y;                       // 0..E/2-1 pair index
    const int c  = kp >> 4;
    const int hw = (kp >> 3) & 1;
    const int v  = kp & 7;
    const float a = pm[(size_t)(2 * kp)     * NN_ + n];
    const float b = pm[(size_t)(2 * kp + 1) * NN_ + n];
    P4[(((size_t)c * 2 + hw) * NN_ + n) * 8 + v] = pack_bf16x2(a, b);
}

// ---------------------------------------------------------------------------
// Kernel 2: GEMM  C[M x N] = A[M x K](f32) * B(K x N, fragment-packed bf16)
// stored TRANSPOSED:  CT[n*M + i]   (bf16 if OUT_BF16 else f32)
// Workgroup: 256 threads = 8 waves. Tile: 256(M) x 128(N), K chunked by 32.
// Each wave computes a 32x128 sub-tile = 2x8 v_wmma_f32_16x16x32_bf16.
// A staged through DOUBLE-BUFFERED LDS with f32->bf16 conversion
// (80 B padded rows, one barrier per chunk).
// ---------------------------------------------------------------------------
#define A_ROW_E  40                         // padded row length in bf16 elems
#define A_BUF_E  (256 * A_ROW_E)            // one buffer, in bf16 elems

template<int OUT_BF16>
__global__ __launch_bounds__(256) void gemm_wmma_kernel(
    const float*    __restrict__ A,
    const unsigned* __restrict__ P4,
    void*           __restrict__ CTv,
    int M, int N, int K)
{
    __shared__ uint4 AsmRaw[2 * A_BUF_E * 2 / 16];   // 2 x 20 KB
    unsigned short* Asm = (unsigned short*)AsmRaw;

    const int tid  = threadIdx.x;
    const int lane = tid & 31;
    const int wave = tid >> 5;
    const int l16  = lane & 15;
    const int hw   = lane >> 4;            // half-wave: 0 or 1

    const int n0   = blockIdx.x * 128;
    const int row0 = blockIdx.y * 256;
    const int nC   = K / 32;

    v8f acc[2][8];
#pragma unroll
    for (int rs = 0; rs < 2; ++rs)
#pragma unroll
        for (int cs = 0; cs < 8; ++cs)
            acc[rs][cs] = (v8f){0.f, 0.f, 0.f, 0.f, 0.f, 0.f, 0.f, 0.f};

    // cooperative A staging coordinates: 8 threads cover one row's 32 K
    const int kt   = tid & 7;              // k-quad within chunk
    const int rowt = tid >> 3;             // row 0..31 (strided by 32, 8 reps)
    const float* Abase = A + (size_t)(row0 + rowt) * K + kt * 4;

    // ---- prologue: stage chunk 0 into buffer 0 ----
#pragma unroll
    for (int j = 0; j < 8; ++j) {
        const float4 f4 = *(const float4*)(Abase + (size_t)j * 32 * K);
        *(uint2*)(&Asm[rowt * A_ROW_E + kt * 4 + j * 32 * A_ROW_E]) =
            make_uint2(pack_bf16x2(f4.x, f4.y), pack_bf16x2(f4.z, f4.w));
    }

    int p = 0;
    for (int c = 0; c < nC; ++c, p ^= 1) {
        __syncthreads();   // staging of buf[p] complete; buf[p^1] fully consumed

        // ---- A fragments from LDS buf[p] (contiguous k order: half-wave hw holds K 16*hw..16*hw+15) ----
        AFrag af[2];
#pragma unroll
        for (int rs = 0; rs < 2; ++rs) {
            const char* ap = (const char*)&Asm[p * A_BUF_E
                                + (wave * 32 + rs * 16 + l16) * A_ROW_E]
                           + hw * 32;                 // half-wave K offset
            af[rs].q[0] = ((const uint4*)ap)[0];      // K {0..7} or {16..23}
            af[rs].q[1] = ((const uint4*)ap)[1];      // K {8..15} or {24..31}
        }

        // ---- B fragments: two contiguous b128 loads per 16-col fragment ----
        BFrag bfr[8];
        const uint4* bp = (const uint4*)(P4 + (((size_t)c * 2 + hw) * N + n0 + l16) * 8);
#pragma unroll
        for (int cs = 0; cs < 8; ++cs) {
            bfr[cs].q[0] = bp[cs * 32];       // 16 cols * 8 dwords = 32 uint4
            bfr[cs].q[1] = bp[cs * 32 + 1];
        }

        // ---- stage chunk c+1 into buf[p^1] (overlaps with WMMAs below) ----
        if (c + 1 < nC) {
            const float* An = Abase + (size_t)(c + 1) * 32;
#pragma unroll
            for (int j = 0; j < 8; ++j) {
                const float4 f4 = *(const float4*)(An + (size_t)j * 32 * K);
                *(uint2*)(&Asm[(p ^ 1) * A_BUF_E + rowt * A_ROW_E + kt * 4
                               + j * 32 * A_ROW_E]) =
                    make_uint2(pack_bf16x2(f4.x, f4.y), pack_bf16x2(f4.z, f4.w));
            }
        }

        // ---- 16 WMMAs ----
#pragma unroll
        for (int rs = 0; rs < 2; ++rs)
#pragma unroll
            for (int cs = 0; cs < 8; ++cs)
                acc[rs][cs] = __builtin_amdgcn_wmma_f32_16x16x32_bf16(
                    false, af[rs].v, false, bfr[cs].v,
                    (short)0, acc[rs][cs], false, false);
    }

    // ---- store C transposed: CT[col * M + row], 8 consecutive rows/lane ----
#pragma unroll
    for (int rs = 0; rs < 2; ++rs)
#pragma unroll
        for (int cs = 0; cs < 8; ++cs) {
            const int col = n0 + cs * 16 + l16;
            const int row = row0 + wave * 32 + rs * 16 + hw * 8;
            CFrag cfr; cfr.v = acc[rs][cs];
            if (OUT_BF16) {
                unsigned short* CT = (unsigned short*)CTv;
                uint4 o;
                o.x = pack_bf16x2(cfr.f[0], cfr.f[1]);
                o.y = pack_bf16x2(cfr.f[2], cfr.f[3]);
                o.z = pack_bf16x2(cfr.f[4], cfr.f[5]);
                o.w = pack_bf16x2(cfr.f[6], cfr.f[7]);
                *(uint4*)(&CT[(size_t)col * M + row]) = o;
            } else {
                float* CT = (float*)CTv;
                *(float4*)(&CT[(size_t)col * M + row])     = cfr.q[0];
                *(float4*)(&CT[(size_t)col * M + row + 4]) = cfr.q[1];
            }
        }
}

// ---------------------------------------------------------------------------
// Kernel 3: layer-1 apply.  One block per sample b (256 threads = 8 waves).
//   h[b,l] = relu( sum_f x[b,f] * (root_lin1[f,l] + D1T[n_b, l*F+f]) )
// ---------------------------------------------------------------------------
__global__ __launch_bounds__(256) void layer1_kernel(
    const float* __restrict__ x, const int* __restrict__ node_idx,
    const unsigned short* __restrict__ D1T, const float* __restrict__ r1,
    float* __restrict__ h)
{
    const int b    = blockIdx.x;
    const int tid  = threadIdx.x;
    const int lane = tid & 31;
    const int wave = tid >> 5;

    __shared__ float xs[FF_];
    xs[tid] = x[(size_t)b * FF_ + tid];
    __syncthreads();

    const int n = node_idx[b];
    const unsigned short* base = D1T + (size_t)n * (L0_ * FF_);

    float xr[8];
#pragma unroll
    for (int q = 0; q < 8; ++q) xr[q] = xs[lane * 8 + q];

    for (int li = 0; li < 16; ++li) {
        const int l = wave * 16 + li;
        const uint4 dv = *(const uint4*)(base + l * FF_ + lane * 8);
        const unsigned dw[4] = {dv.x, dv.y, dv.z, dv.w};
        float acc = 0.f;
#pragma unroll
        for (int q = 0; q < 8; ++q) {
            const unsigned bits = (q & 1) ? (dw[q >> 1] & 0xffff0000u)
                                          : (dw[q >> 1] << 16);
            const float d  = __builtin_bit_cast(float, bits);
            const float rr = r1[(size_t)(lane * 8 + q) * L0_ + l];
            acc += xr[q] * (d + rr);
        }
#pragma unroll
        for (int off = 16; off; off >>= 1) acc += __shfl_xor(acc, off, 32);
        if (lane == 0) h[(size_t)b * L0_ + l] = fmaxf(acc, 0.f);
    }
}

// ---------------------------------------------------------------------------
// Kernel 4: layer-2 apply.  One block per sample b (128 threads).
//   out[b,m] = sum_l h[b,l] * (root_lin2[l,m] + D2T[n_b, m*L0+l])
// ---------------------------------------------------------------------------
__global__ __launch_bounds__(128) void layer2_kernel(
    const float* __restrict__ h, const int* __restrict__ node_idx,
    const float* __restrict__ D2T, const float* __restrict__ r2,
    float* __restrict__ out)
{
    const int b   = blockIdx.x;
    const int tid = threadIdx.x;

    __shared__ float hs[L0_];
    hs[tid] = h[(size_t)b * L0_ + tid];
    __syncthreads();

    const int n  = node_idx[b];
    const int m  = tid >> 3;
    const int lg = tid & 7;
    const float* dbase = D2T + (size_t)n * (L1_ * L0_) + m * L0_ + lg * 16;

    float acc = 0.f;
#pragma unroll
    for (int i = 0; i < 16; ++i) {
        const int l = lg * 16 + i;
        acc += hs[l] * (dbase[i] + r2[l * L1_ + m]);
    }
    acc += __shfl_xor(acc, 1, 32);
    acc += __shfl_xor(acc, 2, 32);
    acc += __shfl_xor(acc, 4, 32);
    if (lg == 0) out[(size_t)b * L1_ + m] = acc;
}

// ---------------------------------------------------------------------------
// Launch
// ---------------------------------------------------------------------------
extern "C" void kernel_launch(void* const* d_in, const int* in_sizes, int n_in,
                              void* d_out, int out_size, void* d_ws, size_t ws_size,
                              hipStream_t stream)
{
    const float* x        = (const float*)d_in[0];
    const int*   node_idx = (const int*)  d_in[1];
    const float* path_mat = (const float*)d_in[2];
    const float* r1       = (const float*)d_in[3];
    const float* r2       = (const float*)d_in[4];
    const float* dm1      = (const float*)d_in[5];
    const float* dm2      = (const float*)d_in[6];
    float* out = (float*)d_out;

    // workspace layout (bytes): 8 MB + 134 MB + 16 MB + 2 MB = ~161 MB
    char* ws = (char*)d_ws;
    const size_t szP4  = (size_t)(EE_ / 2) * NN_ * sizeof(unsigned);
    const size_t szD1T = (size_t)NN_ * (L0_ * FF_) * sizeof(unsigned short);
    const size_t szD2T = (size_t)NN_ * (L1_ * L0_) * sizeof(float);
    unsigned*       P4  = (unsigned*)(ws);
    unsigned short* D1T = (unsigned short*)(ws + szP4);
    float*          D2T = (float*)(ws + szP4 + szD1T);
    float*          hbuf= (float*)(ws + szP4 + szD1T + szD2T);

    // 1) pack path_mat -> WMMA B-fragment-ordered bf16 pairs
    pack_path_kernel<<<dim3(NN_ / 256, EE_ / 2), 256, 0, stream>>>(path_mat, P4);

    // 2) D1T (bf16) = (delta1 as 32768 x 2048) * P, stored transposed
    gemm_wmma_kernel<1><<<dim3(NN_ / 128, (L0_ * FF_) / 256), 256, 0, stream>>>(
        dm1, P4, (void*)D1T, L0_ * FF_, NN_, EE_);

    // 3) D2T (f32) = (delta2 as 2048 x 2048) * P, stored transposed
    gemm_wmma_kernel<0><<<dim3(NN_ / 128, (L1_ * L0_) / 256), 256, 0, stream>>>(
        dm2, P4, (void*)D2T, L1_ * L0_, NN_, EE_);

    // 4) h = relu(x . (r1 + gather(D1T)))
    layer1_kernel<<<BB_, 256, 0, stream>>>(x, node_idx, D1T, r1, hbuf);

    // 5) out = h . (r2 + gather(D2T))
    layer2_kernel<<<BB_, 128, 0, stream>>>(hbuf, node_idx, D2T, r2, out);
}
